// RNN_31456340476020
// MI455X (gfx1250) — matrix operand order repaired and checked by reading, not compiled, not run on hardware
//
#include <hip/hip_runtime.h>

#define T_STEPS 4096
#define BATCH   2048
#define NHID    32

typedef __attribute__((ext_vector_type(16))) _Float16 v16h;
typedef __attribute__((ext_vector_type(2)))  _Float16 h2;
typedef __attribute__((ext_vector_type(8)))  float    v8f;
typedef __attribute__((ext_vector_type(8)))  int      v8i;
typedef __fp16 v2fp16 __attribute__((ext_vector_type(2)));

// Single-instruction f32x2 -> packed f16 (v_cvt_pk_rtz_f16_f32)
static __device__ __forceinline__ unsigned pack2f16(float a, float b) {
    v2fp16 p = __builtin_amdgcn_cvt_pkrtz(a, b);
    return __builtin_bit_cast(unsigned, p);
}

// Single-instruction f32 relu (epilogue only)
static __device__ __forceinline__ float relu1(float x) {
    float r;
    asm("v_max_num_f32 %0, 0, %1" : "=v"(r) : "v"(x));
    return r;
}

// Packed f16 relu: v_pk_max_num_f16 dst, 0, src
static __device__ __forceinline__ unsigned pkrelu(unsigned v) {
    unsigned r;
    asm("v_pk_max_num_f16 %0, 0, %1" : "=v"(r) : "v"(v));
    return r;
}

// Lane-half swap (lane <-> lane^16) in one ds_swizzle_b32, no index VGPR:
// group-of-32 mode, xor_mask=0x10, or_mask=0, and_mask=0x1F -> imm 0x401F
static __device__ __forceinline__ unsigned swap16(unsigned v) {
    return (unsigned)__builtin_amdgcn_ds_swizzle((int)v, 0x401F);
}

static __device__ __forceinline__ v8f wmma_f32_16x16x32_f16(v16h a, v16h b, v8f c) {
    // D = A(16x32 f16) * B(32x16 f16) + C(16x16 f32)
    return __builtin_amdgcn_wmma_f32_16x16x32_f16(
        /*neg_a=*/false, a, /*neg_b=*/false, b,
        /*c_mod=*/(short)0, c, /*reuse_a=*/false, /*reuse_b=*/false);
}

__global__ __launch_bounds__(32)
void rnn_wmma_kernel(const float* __restrict__ x,       // [T, B, 1]
                     const float* __restrict__ hidden,  // [1, B, 32]
                     const float* __restrict__ W_ih,    // [32, 1]
                     const float* __restrict__ W_hh,    // [32, 32]
                     const float* __restrict__ b_ih,    // [32]
                     const float* __restrict__ b_hh,    // [32]
                     const float* __restrict__ W_dec,   // [1, 32]
                     const float* __restrict__ b_dec,   // [1]
                     float* __restrict__ out)           // [T*B] decoded ++ [B*32] h_final
{
    const int lane = threadIdx.x;        // 0..31 (wave32)
    const int col  = lane & 15;          // batch column within tile (also N index)
    const int half = lane >> 4;          // lane half
    const int b0   = blockIdx.x * 16;    // this wave's batch rows [b0, b0+16)

    // ---- Preload A = W_hh in 16-bit A-matrix 16x32 layout (constant over time) ----
    v8i a0i, a1i;
#pragma unroll
    for (int k = 0; k < 8; ++k) {
        const int kb = (half << 4) + (k << 1);
        a0i[k] = (int)pack2f16(W_hh[col * NHID + kb],        W_hh[col * NHID + kb + 1]);
        a1i[k] = (int)pack2f16(W_hh[(16 + col) * NHID + kb], W_hh[(16 + col) * NHID + kb + 1]);
    }
    const v16h A0 = __builtin_bit_cast(v16h, a0i);   // W_hh rows  0..15
    const v16h A1 = __builtin_bit_cast(v16h, a1i);   // W_hh rows 16..31

    // ---- Per-lane constants in D layout: D-tile0 row j -> hid j+8*half, tile1 -> +16 ----
    float wih0[8], wih1[8], bia0[8], bia1[8];
    h2 wdp0[4], wdp1[4];                 // packed f16 W_dec pairs matching ap/dp packing
#pragma unroll
    for (int j = 0; j < 8; ++j) {
        const int h0 = j + 8 * half;
        const int h1 = 16 + h0;
        wih0[j] = W_ih[h0];               wih1[j] = W_ih[h1];
        bia0[j] = b_ih[h0] + b_hh[h0];    bia1[j] = b_ih[h1] + b_hh[h1];
    }
#pragma unroll
    for (int p = 0; p < 4; ++p) {
        const int h0 = 2 * p + 8 * half;
        wdp0[p][0] = (_Float16)W_dec[h0];      wdp0[p][1] = (_Float16)W_dec[h0 + 1];
        wdp1[p][0] = (_Float16)W_dec[16 + h0]; wdp1[p][1] = (_Float16)W_dec[16 + h0 + 1];
    }
    const float bdec = b_dec[0];

    // ---- Initial B operand = hidden.T in 16-bit B-matrix 32x16 layout ----
    v8i bi;
#pragma unroll
    for (int k = 0; k < 8; ++k) {
        const int hb = 16 * half + 2 * k;
        bi[k] = (int)pack2f16(hidden[(b0 + col) * NHID + hb], hidden[(b0 + col) * NHID + hb + 1]);
    }
    v16h B = __builtin_bit_cast(v16h, bi);

    // Raw (pre-relu) WMMA outputs, persist across iterations; relu'd in epilogue for h_final.
    v8f e0 = {}, e1 = {};

    // Pipelined decode state from step t-1 (local half-sum + swizzled other half-sum).
    float    accf  = 0.0f;
    unsigned accsw = 0;

    // Software-prefetch the x scalar one step ahead.
    float nx = x[b0 + col];

    for (int t = 0; t < T_STEPS; ++t) {
        const float xv = nx;
        const int   tn = (t + 1 < T_STEPS) ? (t + 1) : t;
        nx = x[tn * BATCH + b0 + col];

        // C = x_proj.T tiles in D layout (fused as WMMA accumulator)
        v8f c0, c1;
#pragma unroll
        for (int j = 0; j < 8; ++j) {
            c0[j] = fmaf(xv, wih0[j], bia0[j]);
            c1[j] = fmaf(xv, wih1[j], bia1[j]);
        }

        // h_new.T = W_hh @ h.T + x_proj.T   (two M-tiles)
        e0 = wmma_f32_16x16x32_f16(A0, B, c0);
        e1 = wmma_f32_16x16x32_f16(A1, B, c1);

        // ---- pipelined decode tail for step t-1 (independent of e0/e1):
        // fills the WMMA->VALU hazard shadow; its swizzle was issued a full
        // iteration ago, so the dscnt wait is free. t=0 writes a dummy value
        // to row 0 which the t=1 store deterministically overwrites.
        {
            const float acc = accf + __builtin_bit_cast(float, accsw);
            const float ee  = __expf(-(acc + bdec));
            const float y   = __builtin_amdgcn_rcpf(1.0f + ee);
            const int tprev = (t > 0) ? (t - 1) : 0;
            out[tprev * BATCH + b0 + col] = y;   // duplicate halves write same value
        }

        // ---- pack raw f32 pairs to f16, relu in packed f16 domain ----
        unsigned apv[4], dpv[4];
#pragma unroll
        for (int p = 0; p < 4; ++p) {
            apv[p] = pkrelu(pack2f16(e0[2 * p], e0[2 * p + 1]));  // hid (2p,2p+1)+8*half
            dpv[p] = pkrelu(pack2f16(e1[2 * p], e1[2 * p + 1]));  // hid 16+(2p,2p+1)+8*half
        }

        // ---- decode accumulation for this step (consumed next iteration) ----
        h2 acc0 = {}, acc1 = {};
#pragma unroll
        for (int p = 0; p < 4; ++p) {
            acc0 = __builtin_elementwise_fma(__builtin_bit_cast(h2, apv[p]), wdp0[p], acc0);
            acc1 = __builtin_elementwise_fma(__builtin_bit_cast(h2, dpv[p]), wdp1[p], acc1);
        }

        // ---- D -> next B relayout: issue the 4 relayout swizzles first ----
        unsigned m1v[4], swv[4];
#pragma unroll
        for (int p = 0; p < 4; ++p) {
            m1v[p] = half ? dpv[p] : apv[p];           // value this lane keeps locally
            const unsigned m2 = half ? apv[p] : dpv[p]; // value this lane ships across
            swv[p] = swap16(m2);
        }

        // decode half-sum + its swizzle (5th DS op, consumed next iteration)
        const h2 s = acc0 + acc1;                       // v_pk_add_f16
        accf  = (float)s[0] + (float)s[1];
        accsw = swap16(__builtin_bit_cast(unsigned, accf));

        // consume relayout swizzles -> next B
        v8i n;
#pragma unroll
        for (int p = 0; p < 4; ++p) {
            n[p]     = (int)(half ? swv[p] : m1v[p]);
            n[p + 4] = (int)(half ? m1v[p] : swv[p]);
        }
        B = __builtin_bit_cast(v16h, n);
    }

    // ---- epilogue: decode tail for the final step t = T-1 ----
    {
        const float acc = accf + __builtin_bit_cast(float, accsw);
        const float ee  = __expf(-(acc + bdec));
        const float y   = __builtin_amdgcn_rcpf(1.0f + ee);
        out[(T_STEPS - 1) * BATCH + b0 + col] = y;
    }

    // ---- h_final: [B, 32] f32 at offset T*B (relu applied here, full f32) ----
    float* __restrict__ hf = out + (size_t)T_STEPS * BATCH;
#pragma unroll
    for (int j = 0; j < 8; ++j) {
        const int h0 = j + 8 * half;
        hf[(b0 + col) * NHID + h0]      = relu1(e0[j]);
        hf[(b0 + col) * NHID + 16 + h0] = relu1(e1[j]);
    }
}

extern "C" void kernel_launch(void* const* d_in, const int* in_sizes, int n_in,
                              void* d_out, int out_size, void* d_ws, size_t ws_size,
                              hipStream_t stream) {
    const float* x      = (const float*)d_in[0];
    const float* hidden = (const float*)d_in[1];
    const float* W_ih   = (const float*)d_in[2];
    const float* W_hh   = (const float*)d_in[3];
    const float* b_ih   = (const float*)d_in[4];
    const float* b_hh   = (const float*)d_in[5];
    const float* W_dec  = (const float*)d_in[6];
    const float* b_dec  = (const float*)d_in[7];
    float* out = (float*)d_out;

    // One 16-row batch tile (one recurrence chain) per wave32: 128 single-wave
    // blocks -> one wave per SIMD across the device; per-step issue count is
    // minimized since the T=4096 recurrence is the serial bottleneck.
    rnn_wmma_kernel<<<BATCH / 16, 32, 0, stream>>>(
        x, hidden, W_ih, W_hh, b_ih, b_hh, W_dec, b_dec, out);
}
